// EdgeScoreDotProductTransformer_4698694221869
// MI455X (gfx1250) — matrix operand order repaired and checked by reading, not compiled, not run on hardware
//
#include <hip/hip_runtime.h>
#include <hip/hip_bf16.h>

// ---------------------------------------------------------------------------
// EdgeScoreDotProductTransformer for MI455X (gfx1250, wave32, WMMA)
//
//   Q = h_src @ Wq  (50000x256 @ 256x256)   -- bf16-split WMMA GEMM
//   K = h_dst @ Wk  (50000x256 @ 256x256)   -- bf16-split WMMA GEMM
//   score[e,h] = dot(Q[src[e],h,:], K[dst[e],h,:]) / sqrt(32) + (efeat@Be)[e,h]
//
// f32 precision is recovered via the hi/lo bf16 split: x = hi + lo with
// a*b ~= ah*bh + al*bh + ah*bl  (three v_wmma_f32_16x16x32_bf16, f32 accum).
// ---------------------------------------------------------------------------

typedef __attribute__((ext_vector_type(16))) __bf16 v16bf;
typedef __attribute__((ext_vector_type(8)))  float  v8f;

#define NSRC   50000
#define DNODE  256
#define NEDGE  800000
#define NHEAD  8
#define DHEAD  32
#define DEDGE  32

// Fragment store per W matrix: 16 n-tiles * 8 k-chunks * 32 lanes * 16 bf16
#define FRAG_ELEMS 65536   // = 256*256

// ---------------------------------------------------------------------------
// Kernel A: repack Wq/Wk (256x256 f32, row-major [k][n]) into B-fragment
// order as bf16 hi/lo.  This code's own contiguous-k layout (not the ISA table order) for the
// B operand (32x16) has: lane n (n=0..15) holds K=kc+0..15 of column n,
// lane 16+n holds K=kc+16..31 of column n, K ascending through the 8 VGPRs.
// So lane data = 16 *consecutive K* values of one column -> the GEMM can
// load its B fragment as one contiguous 32-byte vector.
// ---------------------------------------------------------------------------
__global__ void build_wfrag_kernel(const float* __restrict__ Wq,
                                   const float* __restrict__ Wk,
                                   __bf16* __restrict__ frag) {
    int g = blockIdx.x * blockDim.x + threadIdx.x;   // 0 .. 8191
    int which = g >> 12;                             // 0 = Wq, 1 = Wk
    int rem   = g & 4095;                            // 16*8*32 tasks per matrix
    int t     = rem >> 8;                            // n-tile   0..15
    int c     = (rem >> 5) & 7;                      // k-chunk  0..7
    int l     = rem & 31;                            // lane     0..31

    const float* W = which ? Wk : Wq;
    __bf16* hi = frag + which * (2 * FRAG_ELEMS) + (((t * 8) + c) * 32 + l) * 16;
    __bf16* lo = hi + FRAG_ELEMS;

    int n     = t * 16 + (l & 15);
    int kbase = c * 32 + (l >> 4) * 16;
#pragma unroll
    for (int j = 0; j < 16; ++j) {
        float  x = W[(kbase + j) * DNODE + n];
        __bf16 h = (__bf16)x;
        hi[j] = h;
        lo[j] = (__bf16)(x - (float)h);
    }
}

// ---------------------------------------------------------------------------
// Kernel B: C(50000x256) = A(50000x256) @ W(256x256) with bf16-split WMMA.
// block = 128 threads (4 waves).  blockIdx.x = M-tile (16 rows),
// blockIdx.y selects (h_src,Wq)->Q or (h_dst,Wk)->K.
// Each wave owns 4 N-tiles of 16 columns; 8 K-chunks of 32; 3 WMMAs/chunk/tile.
// A fragment (contiguous k order, same as the B fragments): lane m=l&15, half=l>>4;
// elements 0..15 -> K = c*32 + half*16 + (0..15)
// i.e. one contiguous 64B load from the row-major A row.
// ---------------------------------------------------------------------------
__global__ void __launch_bounds__(128)
qk_gemm_kernel(const float* __restrict__ hsrc,
               const float* __restrict__ hdst,
               float* __restrict__ Qout,
               float* __restrict__ Kout,
               const __bf16* __restrict__ wfrag) {
    const int mtile = blockIdx.x;
    const int which = blockIdx.y;                    // 0 -> Q, 1 -> K
    const float* A = which ? hdst : hsrc;
    float*       C = which ? Kout : Qout;
    const __bf16* whi = wfrag + which * (2 * FRAG_ELEMS);

    const int tid  = threadIdx.x;
    const int wave = tid >> 5;                       // 0..3
    const int lane = tid & 31;
    const int half = lane >> 4;
    const int mloc = lane & 15;

    const float* arow = A + (mtile * 16 + mloc) * DNODE;

    v8f acc[4] = {};                                 // 4 N-tiles of 16x16 f32

#pragma unroll
    for (int c = 0; c < 8; ++c) {                    // K chunks of 32
        // ---- A fragment: one contiguous 64B read, then hi/lo split ----
        const float* ap = arow + c * 32 + half * 16;
        float f[16];
        *(float4*)(f + 0)  = *(const float4*)(ap + 0);
        *(float4*)(f + 4)  = *(const float4*)(ap + 4);
        *(float4*)(f + 8)  = *(const float4*)(ap + 8);
        *(float4*)(f + 12) = *(const float4*)(ap + 12);

        v16bf ahi, alo;
#pragma unroll
        for (int i = 0; i < 16; ++i) {
            __bf16 h = (__bf16)f[i];
            ahi[i] = h;
            alo[i] = (__bf16)(f[i] - (float)h);
        }

        // ---- 4 N-tiles: contiguous 32B fragment loads + 3 WMMAs each ----
#pragma unroll
        for (int t = 0; t < 4; ++t) {
            const int nt = wave * 4 + t;
            const __bf16* bp = whi + (((nt * 8) + c) * 32 + lane) * 16;
            v16bf bhi = *(const v16bf*)bp;
            v16bf blo = *(const v16bf*)(bp + FRAG_ELEMS);

            acc[t] = __builtin_amdgcn_wmma_f32_16x16x32_bf16(
                false, ahi, false, bhi, (short)0, acc[t], false, false);
            acc[t] = __builtin_amdgcn_wmma_f32_16x16x32_bf16(
                false, alo, false, bhi, (short)0, acc[t], false, false);
            acc[t] = __builtin_amdgcn_wmma_f32_16x16x32_bf16(
                false, ahi, false, blo, (short)0, acc[t], false, false);
        }
    }

    // ---- store: C/D layout lane n = l&15; VGPR v -> row v + half*8 ----
#pragma unroll
    for (int t = 0; t < 4; ++t) {
        const int n0 = (wave * 4 + t) * 16 + (lane & 15);
#pragma unroll
        for (int v = 0; v < 8; ++v) {
            const int m = mtile * 16 + half * 8 + v;
            C[m * DNODE + n0] = acc[t][v];
        }
    }
}

// ---------------------------------------------------------------------------
// Kernel C: per-edge scores.  One lane per (edge, head): 8 lanes/edge read
// contiguous 128B head slices of Q[src] and K[dst] (L2-resident: Q+K =
// 102.4MB < 192MB L2), dot over 32 dims, scale, add efeat@Be bias with Be
// staged in LDS.
// ---------------------------------------------------------------------------
__global__ void __launch_bounds__(256)
edge_score_kernel(const float* __restrict__ Q,
                  const float* __restrict__ K,
                  const float* __restrict__ efeat,
                  const int* __restrict__ src_idx,
                  const int* __restrict__ dst_idx,
                  const float* __restrict__ Be,
                  float* __restrict__ out) {
    __shared__ float sBe[DEDGE * NHEAD];             // 32x8 = 256 floats
    const int tid = threadIdx.x;
    sBe[tid] = Be[tid];
    __syncthreads();

    const int gid = blockIdx.x * 256 + tid;          // grid sized exactly E*H
    const int e = gid >> 3;
    const int h = gid & 7;

    const int s = src_idx[e];
    const int d = dst_idx[e];
    const float4* qp = (const float4*)(Q + s * DNODE + h * DHEAD);
    const float4* kp = (const float4*)(K + d * DNODE + h * DHEAD);

    float acc = 0.0f;
#pragma unroll
    for (int i = 0; i < 8; ++i) {
        float4 q = qp[i];
        float4 k = kp[i];
        acc += q.x * k.x + q.y * k.y + q.z * k.z + q.w * k.w;
    }
    acc *= 0.17677669529663687f;                     // 1/sqrt(32)

    const float4* ep = (const float4*)(efeat + e * DEDGE);
#pragma unroll
    for (int i = 0; i < 8; ++i) {
        float4 f = ep[i];
        acc += f.x * sBe[(i * 4 + 0) * NHEAD + h];
        acc += f.y * sBe[(i * 4 + 1) * NHEAD + h];
        acc += f.z * sBe[(i * 4 + 2) * NHEAD + h];
        acc += f.w * sBe[(i * 4 + 3) * NHEAD + h];
    }

    out[e * NHEAD + h] = acc;
}

// ---------------------------------------------------------------------------
extern "C" void kernel_launch(void* const* d_in, const int* in_sizes, int n_in,
                              void* d_out, int out_size, void* d_ws, size_t ws_size,
                              hipStream_t stream) {
    const float* h_src   = (const float*)d_in[0];
    const float* h_dst   = (const float*)d_in[1];
    const float* efeat   = (const float*)d_in[2];
    const int*   src_idx = (const int*)  d_in[3];
    const int*   dst_idx = (const int*)  d_in[4];
    const float* Wq      = (const float*)d_in[5];
    const float* Wk      = (const float*)d_in[6];
    const float* Be      = (const float*)d_in[7];
    float*       out     = (float*)d_out;

    // Workspace layout (all offsets 256B aligned):
    //   [0, 51.2MB)      Q  f32  (50000x256)
    //   [51.2, 102.4MB)  K  f32  (50000x256)
    //   [102.4MB, +512KB) Wq_hi | Wq_lo | Wk_hi | Wk_lo  bf16 fragments
    char*   ws   = (char*)d_ws;
    float*  Qf   = (float*)ws;
    float*  Kf   = Qf + (size_t)NSRC * DNODE;
    __bf16* frag = (__bf16*)(ws + (size_t)2 * NSRC * DNODE * sizeof(float));

    // A: repack W matrices into WMMA fragment order (8192 lane-tasks)
    build_wfrag_kernel<<<32, 256, 0, stream>>>(Wq, Wk, frag);

    // B: Q and K projections via bf16-split WMMA (grid.y: 0->Q, 1->K)
    dim3 ggemm(NSRC / 16, 2);
    qk_gemm_kernel<<<ggemm, 128, 0, stream>>>(h_src, h_dst, Qf, Kf, frag);

    // C: edge scores (E*H = 6.4M lanes, exactly 25000 blocks of 256)
    edge_score_kernel<<<(NEDGE * NHEAD) / 256, 256, 0, stream>>>(
        Qf, Kf, efeat, src_idx, dst_idx, Be, out);
}
